// DrumTransformer_17248588660826
// MI455X (gfx1250) — matrix operand order repaired and checked by reading, not compiled, not run on hardware
//
#include <hip/hip_runtime.h>
#include <hip/hip_bf16.h>
#include <math.h>

// ---------------------------------------------------------------------------
// DrumTransformer forward on MI455X (gfx1250, wave32).
// GEMMs (the ~0.3 TFLOP hot path) use V_WMMA_F32_16X16X32_F16 with a
// software-pipelined staging loop: after tile i is stored regs->LDS, tile i+1's
// global_load_b128s are issued to overlap tile i's WMMAs, and all fragment ds_load_b128s are issued
// before the back-to-back WMMA chain. Linear attention (O(L*D^2), negligible
// FLOPs) uses LDS/VALU kernels.
// ---------------------------------------------------------------------------

typedef __attribute__((ext_vector_type(16))) _Float16 v16h;
typedef __attribute__((ext_vector_type(8)))  _Float16 v8h;
typedef __attribute__((ext_vector_type(4)))  _Float16 v4h;
typedef __attribute__((ext_vector_type(8)))  float    v8f;

#define BB    4
#define LL    512
#define DM    512
#define NH    8
#define DH    64
#define DFF   2048
#define MROWS (BB*LL)          // 2048 token rows
#define DEMB_W 1216            // sum of decoder embedding widths

// ======================== WMMA GEMM ========================
// C[M,N] = A[M,K] @ B[K,N] + bias[N] (+ Res[M,N])
// Block tile 128x64, K-step 32, 256 threads = 8 waves.
// Wave w computes rows [w*16, w*16+16) x all 64 cols -> 4 wmma accumulators.
// LDS layout: As row-major (rows padded to 40 halves, 80B stride);
//             Bt stored TRANSPOSED (n-major) so B fragments are contiguous.
template<bool ADD_RES>
__global__ __launch_bounds__(256)
void gemm_f16wmma(const float* __restrict__ A, const float* __restrict__ Bm,
                  const float* __restrict__ bias, const float* __restrict__ Res,
                  float* __restrict__ C, int M, int N, int K)
{
    __shared__ _Float16 As[128][40];   // [m][k], 80B row stride (16B aligned)
    __shared__ _Float16 Bt[64][40];    // [n][k], transposed B tile

    const int tid  = threadIdx.x;
    const int wave = tid >> 5;
    const int lane = tid & 31;
    const int m0   = blockIdx.y * 128;
    const int n0   = blockIdx.x * 64;

    v8f acc[4];
#pragma unroll
    for (int i = 0; i < 4; ++i)
#pragma unroll
        for (int j = 0; j < 8; ++j) acc[i][j] = 0.0f;

    // ---- software pipeline registers: tile staged in VGPRs ----
    float4 fa[4], fb[2];
    // prologue: load tile kk=0 (all loads issued before any use -> MLP)
#pragma unroll
    for (int t = 0; t < 4; ++t) {
        int idx = tid + t * 256;               // 0..1023 float4 chunks of A
        int r = idx >> 3, c4 = idx & 7;
        fa[t] = *(const float4*)&A[(size_t)(m0 + r) * K + c4 * 4];
    }
#pragma unroll
    for (int t = 0; t < 2; ++t) {
        int idx = tid + t * 256;               // 0..511 float4 chunks of B
        int r = idx >> 4, c4 = idx & 15;
        fb[t] = *(const float4*)&Bm[(size_t)r * N + n0 + c4 * 4];
    }

    for (int kk = 0; kk < K; kk += 32) {
        // ---- store staged tile regs -> LDS (f32->f16) ----
#pragma unroll
        for (int t = 0; t < 4; ++t) {
            int idx = tid + t * 256;
            int r = idx >> 3, c4 = idx & 7;
            v4h h; h[0] = (_Float16)fa[t].x; h[1] = (_Float16)fa[t].y;
                   h[2] = (_Float16)fa[t].z; h[3] = (_Float16)fa[t].w;
            *(v4h*)&As[r][c4 * 4] = h;         // 8B aligned -> ds_store_b64
        }
#pragma unroll
        for (int t = 0; t < 2; ++t) {
            int idx = tid + t * 256;
            int r = idx >> 4, c4 = idx & 15;   // k row, 4-col chunk
            Bt[c4 * 4 + 0][r] = (_Float16)fb[t].x;
            Bt[c4 * 4 + 1][r] = (_Float16)fb[t].y;
            Bt[c4 * 4 + 2][r] = (_Float16)fb[t].z;
            Bt[c4 * 4 + 3][r] = (_Float16)fb[t].w;
        }
        __syncthreads();

        // ---- issue next tile's global loads; they overlap the WMMAs below ----
        if (kk + 32 < K) {
            const int k2 = kk + 32;
#pragma unroll
            for (int t = 0; t < 4; ++t) {
                int idx = tid + t * 256;
                int r = idx >> 3, c4 = idx & 7;
                fa[t] = *(const float4*)&A[(size_t)(m0 + r) * K + k2 + c4 * 4];
            }
#pragma unroll
            for (int t = 0; t < 2; ++t) {
                int idx = tid + t * 256;
                int r = idx >> 4, c4 = idx & 15;
                fb[t] = *(const float4*)&Bm[(size_t)(k2 + r) * N + n0 + c4 * 4];
            }
            if (kk + 64 < K) {                 // prefetch tile after next into L2
                __builtin_prefetch(&A[(size_t)(m0 + (tid >> 1)) * K + kk + 64 + (tid & 1) * 16], 0, 1);
                __builtin_prefetch(&Bm[(size_t)(kk + 64 + (tid >> 3)) * N + n0 + (tid & 7) * 8], 0, 1);
            }
        }

        // ---- all fragment LDS loads first, then back-to-back WMMAs ----
        const int mr = wave * 16 + (lane & 15);
        const int hi = lane >> 4;
        v8h a0 = *(const v8h*)&As[mr][hi * 8];          // k = hi*8 .. +7
        v8h a1 = *(const v8h*)&As[mr][16 + hi * 8];     // k = 16+hi*8 ..
        v16h af = __builtin_shufflevector(a0, a1, 0, 1, 2, 3, 4, 5, 6, 7,
                                                  8, 9, 10, 11, 12, 13, 14, 15);
        v16h bf[4];
#pragma unroll
        for (int nb = 0; nb < 4; ++nb) {
            const int nc = nb * 16 + (lane & 15);
            v8h b0 = *(const v8h*)&Bt[nc][hi * 8];      // k = hi*8 .. +7
            v8h b1 = *(const v8h*)&Bt[nc][16 + hi * 8]; // k = 16+hi*8 .. +7
            bf[nb] = __builtin_shufflevector(b0, b1, 0, 1, 2, 3, 4, 5, 6, 7,
                                                     8, 9, 10, 11, 12, 13, 14, 15);
        }
#pragma unroll
        for (int nb = 0; nb < 4; ++nb)
            acc[nb] = __builtin_amdgcn_wmma_f32_16x16x32_f16(
                false, af, false, bf[nb], (short)0, acc[nb], false, false);
        __syncthreads();
    }

    // ---- epilogue: bias (+ residual) ----
#pragma unroll
    for (int nb = 0; nb < 4; ++nb) {
        int col = n0 + nb * 16 + (lane & 15);
        float bv = bias[col];
#pragma unroll
        for (int r = 0; r < 8; ++r) {
            int row = m0 + wave * 16 + (lane >> 4) * 8 + r;
            float v = acc[nb][r] + bv;
            if (ADD_RES) v += Res[(size_t)row * N + col];
            C[(size_t)row * N + col] = v;
        }
    }
}

// ======================== LayerNorm (row = 512) ========================
__global__ __launch_bounds__(256)
void layernorm_k(const float* __restrict__ X, const float* __restrict__ g,
                 const float* __restrict__ b, float* __restrict__ O)
{
    __shared__ float red[256];
    const int row = blockIdx.x, tid = threadIdx.x;
    const float* x = X + (size_t)row * DM;
    float v0 = x[tid], v1 = x[tid + 256];
    red[tid] = v0 + v1;
    __syncthreads();
    for (int st = 128; st > 0; st >>= 1) { if (tid < st) red[tid] += red[tid + st]; __syncthreads(); }
    float mean = red[0] * (1.0f / DM);
    __syncthreads();
    float d0 = v0 - mean, d1 = v1 - mean;
    red[tid] = d0 * d0 + d1 * d1;
    __syncthreads();
    for (int st = 128; st > 0; st >>= 1) { if (tid < st) red[tid] += red[tid + st]; __syncthreads(); }
    float inv = rsqrtf(red[0] * (1.0f / DM) + 1e-5f);
    float* o = O + (size_t)row * DM;
    o[tid]       = d0 * inv * g[tid]       + b[tid];
    o[tid + 256] = d1 * inv * g[tid + 256] + b[tid + 256];
}

// ======================== elementwise ========================
__global__ void fm_k(float* X, int n) {           // elu(x)+1
    int i = blockIdx.x * blockDim.x + threadIdx.x;
    if (i < n) { float x = X[i]; X[i] = x > 0.0f ? x + 1.0f : expf(x); }
}
__global__ void gelu_k(float* X, int n) {         // exact gelu
    int i = blockIdx.x * blockDim.x + threadIdx.x;
    if (i < n) { float x = X[i]; X[i] = 0.5f * x * (1.0f + erff(x * 0.70710678118654752f)); }
}

// ======================== embeddings ========================
__global__ void enc_embed_k(const int* __restrict__ en_x,
                            const float* __restrict__ emb_bb,
                            const float* __restrict__ emb_bin,
                            float* __restrict__ EB, float* __restrict__ EBIN)
{
    int idx = blockIdx.x * blockDim.x + threadIdx.x;
    if (idx >= MROWS * 256) return;
    int row = idx >> 8, c = idx & 255;
    EB[idx]   = emb_bb [en_x[row * 2 + 0] * 256 + c] * 16.0f;   // sqrt(256)
    EBIN[idx] = emb_bin[en_x[row * 2 + 1] * 256 + c] * 16.0f;
}

__global__ void enc_combine_k(const float* __restrict__ T0, const float* __restrict__ T1,
                              const float* __restrict__ pos_emb,
                              const unsigned char* __restrict__ disBE,
                              const unsigned char* __restrict__ disPE,
                              float* __restrict__ X)
{
    int idx = blockIdx.x * blockDim.x + threadIdx.x;
    if (idx >= MROWS * DM) return;
    int row = idx >> 9, c = idx & 511, l = row & 511;
    float v = T1[idx];
    if (!disBE[0]) v += T0[idx];
    if (!disPE[0]) v += pos_emb[l * DM + c] * 22.62741699796952f;  // sqrt(512)
    X[idx] = v;
}

__global__ void dec_embed_k(const int* __restrict__ de_x,
                            const float* __restrict__ e_type, const float* __restrict__ e_bb,
                            const float* __restrict__ e_bd,   const float* __restrict__ e_pitch,
                            const float* __restrict__ e_dur,  const float* __restrict__ e_vel,
                            const float* __restrict__ e_od,   float* __restrict__ DEMB)
{
    int idx = blockIdx.x * blockDim.x + threadIdx.x;
    if (idx >= MROWS * DEMB_W) return;
    int row = idx / DEMB_W, c = idx - row * DEMB_W;
    const int* dx = de_x + row * 8;
    float v;
    if      (c < 64)   v = e_type [dx[0] * 64  +  c        ] * 8.0f;
    else if (c < 320)  v = e_bb   [dx[1] * 256 + (c - 64)  ] * 16.0f;
    else if (c < 384)  v = e_bd   [dx[2] * 64  + (c - 320) ] * 8.0f;
    else if (c < 896)  v = e_pitch[dx[3] * 512 + (c - 384) ] * 22.62741699796952f;
    else if (c < 1024) v = e_dur  [dx[4] * 128 + (c - 896) ] * 11.313708498984761f;
    else if (c < 1152) v = e_vel  [dx[5] * 128 + (c - 1024)] * 11.313708498984761f;
    else               v = e_od   [dx[6] * 64  + (c - 1152)] * 8.0f;
    DEMB[idx] = v;
}

__global__ void beat_pe_k(const int* __restrict__ de_x, float* __restrict__ Y)
{
    int idx = blockIdx.x * blockDim.x + threadIdx.x;
    if (idx >= MROWS * DM) return;
    int row = idx >> 9, c = idx & 511;
    float ang = (float)de_x[row * 8 + 7] *
                expf(-9.210340371976184f * (float)(c & ~1) * (1.0f / 512.0f));
    Y[idx] += (c & 1) ? cosf(ang) : sinf(ang);
}

// ======================== non-causal linear attention ========================
// One workgroup per (b,h). Phase 1: KV[64][64] = K^T V, Ksum[64] in LDS.
// Phase 2: out[l] = (q_l @ KV) / (q_l . Ksum + eps).
__global__ __launch_bounds__(256)
void linattn_k(const float* __restrict__ Q, const float* __restrict__ Kf,
               const float* __restrict__ V, float* __restrict__ O, int Lkv)
{
    __shared__ float Ks[32][64], Vs[32][64];
    __shared__ float KV[64][64];
    __shared__ float Ksum[64];
    const int bh = blockIdx.x, b = bh >> 3, h = bh & 7;
    const int tid = threadIdx.x;
    const int d = tid >> 2, mb = (tid & 3) * 16;

    float acc[16];
#pragma unroll
    for (int j = 0; j < 16; ++j) acc[j] = 0.0f;
    float ks = 0.0f;

    const size_t baseKV = ((size_t)b * Lkv) * DM + h * DH;
    for (int l0 = 0; l0 < Lkv; l0 += 32) {
        __syncthreads();
#pragma unroll
        for (int t = 0; t < 8; ++t) {
            int idx = tid + t * 256;
            int r = idx >> 6, c = idx & 63;
            size_t off = baseKV + (size_t)(l0 + r) * DM + c;
            Ks[r][c] = Kf[off];
            Vs[r][c] = V[off];
        }
        __syncthreads();
#pragma unroll 4
        for (int r = 0; r < 32; ++r) {
            float kval = Ks[r][d];
            if ((tid & 3) == 0) ks += kval;
#pragma unroll
            for (int j = 0; j < 16; ++j) acc[j] += kval * Vs[r][mb + j];
        }
    }
#pragma unroll
    for (int j = 0; j < 16; ++j) KV[d][mb + j] = acc[j];
    if ((tid & 3) == 0) Ksum[d] = ks;
    __syncthreads();

    const size_t baseQ = ((size_t)b * LL) * DM + h * DH;
    for (int p = 0; p < LL / 256; ++p) {
        int l = tid + p * 256;
        float q[64];
#pragma unroll
        for (int dd = 0; dd < 64; ++dd) q[dd] = Q[baseQ + (size_t)l * DM + dd];
        float den = 1e-6f;
#pragma unroll
        for (int dd = 0; dd < 64; ++dd) den += q[dd] * Ksum[dd];
        float inv = 1.0f / den;
        for (int m = 0; m < 64; ++m) {
            float s = 0.0f;
#pragma unroll
            for (int dd = 0; dd < 64; ++dd) s += q[dd] * KV[dd][m];
            O[baseQ + (size_t)l * DM + m] = s * inv;
        }
    }
}

// ======================== causal linear attention ========================
// One workgroup (64 threads) per (b,h). Thread t owns KV column m=t (64 regs).
__global__ __launch_bounds__(64)
void causal_linattn_k(const float* __restrict__ Q, const float* __restrict__ K,
                      const float* __restrict__ V, float* __restrict__ O)
{
    __shared__ float ksh[64], vsh[64], qsh[64], ksum[64];
    const int bh = blockIdx.x, b = bh >> 3, h = bh & 7;
    const int t = threadIdx.x;
    float kv[64];
#pragma unroll
    for (int d = 0; d < 64; ++d) kv[d] = 0.0f;
    ksum[t] = 0.0f;
    __syncthreads();

    const size_t base = ((size_t)b * LL) * DM + h * DH;
    for (int l = 0; l < LL; ++l) {
        size_t off = base + (size_t)l * DM + t;
        ksh[t] = K[off]; vsh[t] = V[off]; qsh[t] = Q[off];
        __syncthreads();
        ksum[t] += ksh[t];
        __syncthreads();
        float v = vsh[t];
        float num = 0.0f, den = 1e-6f;
#pragma unroll
        for (int d = 0; d < 64; ++d) {
            kv[d] += ksh[d] * v;
            float q = qsh[d];
            num += q * kv[d];
            den += q * ksum[d];
        }
        O[off] = num / den;
        __syncthreads();
    }
}

// ======================== output projection [512 -> 4] ========================
__global__ void proj_type_k(const float* __restrict__ Y, const float* __restrict__ W,
                            const float* __restrict__ b, float* __restrict__ O)
{
    int idx = blockIdx.x * blockDim.x + threadIdx.x;
    if (idx >= MROWS * 4) return;
    int row = idx >> 2, t = idx & 3;
    float s = b[t];
    for (int c = 0; c < DM; ++c) s += Y[(size_t)row * DM + c] * W[c * 4 + t];
    O[idx] = s;
}

// ======================== host side ========================
extern "C" void kernel_launch(void* const* d_in, const int* in_sizes, int n_in,
                              void* d_out, int out_size, void* d_ws, size_t ws_size,
                              hipStream_t stream)
{
    (void)in_sizes; (void)n_in; (void)out_size; (void)ws_size;
    // Input order: setup_inputs() dict order (en_x, de_x, disable_BE, disable_PE),
    // then params flattened as a JAX pytree (nested dicts in sorted-key order).
    const int* en_x = (const int*)d_in[0];
    const int* de_x = (const int*)d_in[1];
    const unsigned char* disBE = (const unsigned char*)d_in[2];
    const unsigned char* disPE = (const unsigned char*)d_in[3];
    const float* bin_W    = (const float*)d_in[4];
    const float* bin_b    = (const float*)d_in[5];
    const float* dec_in_W = (const float*)d_in[6];
    const float* dec_in_b = (const float*)d_in[7];
    // dec_layers [8,...] sorted: W1,W2,Wk,Wk2,Wo,Wo2,Wq,Wq2,Wv,Wv2,b1,b2,bk,bk2,
    //                            bo,bo2,bq,bq2,bv,bv2,ln1_b,ln1_g,ln2_b,ln2_g,ln3_b,ln3_g
    const float* dW1  = (const float*)d_in[8];
    const float* dW2  = (const float*)d_in[9];
    const float* dWk  = (const float*)d_in[10];
    const float* dWk2 = (const float*)d_in[11];
    const float* dWo  = (const float*)d_in[12];
    const float* dWo2 = (const float*)d_in[13];
    const float* dWq  = (const float*)d_in[14];
    const float* dWq2 = (const float*)d_in[15];
    const float* dWv  = (const float*)d_in[16];
    const float* dWv2 = (const float*)d_in[17];
    const float* db1  = (const float*)d_in[18];
    const float* db2  = (const float*)d_in[19];
    const float* dbk  = (const float*)d_in[20];
    const float* dbk2 = (const float*)d_in[21];
    const float* dbo  = (const float*)d_in[22];
    const float* dbo2 = (const float*)d_in[23];
    const float* dbq  = (const float*)d_in[24];
    const float* dbq2 = (const float*)d_in[25];
    const float* dbv  = (const float*)d_in[26];
    const float* dbv2 = (const float*)d_in[27];
    const float* dln1b = (const float*)d_in[28];
    const float* dln1g = (const float*)d_in[29];
    const float* dln2b = (const float*)d_in[30];
    const float* dln2g = (const float*)d_in[31];
    const float* dln3b = (const float*)d_in[32];
    const float* dln3g = (const float*)d_in[33];
    const float* emb_barbeat       = (const float*)d_in[34];
    const float* emb_beat_density  = (const float*)d_in[35];
    const float* emb_binary        = (const float*)d_in[36];
    const float* emb_duration      = (const float*)d_in[37];
    const float* emb_onset_density = (const float*)d_in[38];
    const float* emb_pitch         = (const float*)d_in[39];
    const float* emb_type          = (const float*)d_in[40];
    const float* emb_velocity      = (const float*)d_in[41];
    const float* enc_in_W = (const float*)d_in[42];
    const float* enc_in_b = (const float*)d_in[43];
    // enc_layers [8,...] sorted: W1,W2,Wk,Wo,Wq,Wv,b1,b2,bk,bo,bq,bv,ln1_b,ln1_g,ln2_b,ln2_g
    const float* eW1  = (const float*)d_in[44];
    const float* eW2  = (const float*)d_in[45];
    const float* eWk  = (const float*)d_in[46];
    const float* eWo  = (const float*)d_in[47];
    const float* eWq  = (const float*)d_in[48];
    const float* eWv  = (const float*)d_in[49];
    const float* eb1  = (const float*)d_in[50];
    const float* eb2  = (const float*)d_in[51];
    const float* ebk  = (const float*)d_in[52];
    const float* ebo  = (const float*)d_in[53];
    const float* ebq  = (const float*)d_in[54];
    const float* ebv  = (const float*)d_in[55];
    const float* eln1b = (const float*)d_in[56];
    const float* eln1g = (const float*)d_in[57];
    const float* eln2b = (const float*)d_in[58];
    const float* eln2g = (const float*)d_in[59];
    const float* pos_emb = (const float*)d_in[60];
    const float* proj_W  = (const float*)d_in[61];
    const float* proj_b  = (const float*)d_in[62];

    // ---- workspace carve-up (~72 MB of f32) ----
    float* ws = (float*)d_ws;
    size_t off = 0;
    auto alloc = [&](size_t n) { float* p = ws + off; off += n; return p; };
    float* EB   = alloc((size_t)MROWS * 256);
    float* EBIN = alloc((size_t)MROWS * 256);
    float* T0   = alloc((size_t)MROWS * DM);
    float* T1   = alloc((size_t)MROWS * DM);
    float* X    = alloc((size_t)MROWS * DM);    // encoder activation -> memory
    float* Y    = alloc((size_t)MROWS * DM);    // decoder activation
    float* Qb   = alloc((size_t)MROWS * DM);
    float* Kb   = alloc((size_t)MROWS * DM);
    float* Vb   = alloc((size_t)MROWS * DM);
    float* Ab   = alloc((size_t)MROWS * DM);
    float* Rb   = alloc((size_t)MROWS * DM);
    float* Fb   = alloc((size_t)MROWS * DFF);
    float* DEMBp= alloc((size_t)MROWS * DEMB_W);

    auto gemm = [&](const float* A, const float* Bm, const float* bias,
                    const float* Res, float* C, int M, int N, int K) {
        dim3 grid(N / 64, M / 128);
        if (Res) gemm_f16wmma<true><<<grid, 256, 0, stream>>>(A, Bm, bias, Res, C, M, N, K);
        else     gemm_f16wmma<false><<<grid, 256, 0, stream>>>(A, Bm, bias, nullptr, C, M, N, K);
    };
    auto elt = [&](int n) { return dim3((n + 255) / 256); };

    // ==================== encoder ====================
    enc_embed_k<<<elt(MROWS * 256), 256, 0, stream>>>(en_x, emb_barbeat, emb_binary, EB, EBIN);
    gemm(EB,   enc_in_W, enc_in_b, nullptr, T0, MROWS, DM, 256);
    gemm(EBIN, bin_W,    bin_b,    nullptr, T1, MROWS, DM, 256);
    enc_combine_k<<<elt(MROWS * DM), 256, 0, stream>>>(T0, T1, pos_emb, disBE, disPE, X);

    for (int i = 0; i < 8; ++i) {
        const size_t wS = (size_t)i * DM * DM, bS = (size_t)i * DM;
        const size_t w1S = (size_t)i * DM * DFF, b1S = (size_t)i * DFF;
        gemm(X, eWq + wS, ebq + bS, nullptr, Qb, MROWS, DM, DM);
        gemm(X, eWk + wS, ebk + bS, nullptr, Kb, MROWS, DM, DM);
        gemm(X, eWv + wS, ebv + bS, nullptr, Vb, MROWS, DM, DM);
        fm_k<<<elt(MROWS * DM), 256, 0, stream>>>(Qb, MROWS * DM);
        fm_k<<<elt(MROWS * DM), 256, 0, stream>>>(Kb, MROWS * DM);
        linattn_k<<<BB * NH, 256, 0, stream>>>(Qb, Kb, Vb, Ab, LL);
        gemm(Ab, eWo + wS, ebo + bS, X, Rb, MROWS, DM, DM);
        layernorm_k<<<MROWS, 256, 0, stream>>>(Rb, eln1g + bS, eln1b + bS, X);
        gemm(X, eW1 + w1S, eb1 + b1S, nullptr, Fb, MROWS, DFF, DM);
        gelu_k<<<elt(MROWS * DFF), 256, 0, stream>>>(Fb, MROWS * DFF);
        gemm(Fb, eW2 + w1S, eb2 + bS, X, Rb, MROWS, DM, DFF);
        layernorm_k<<<MROWS, 256, 0, stream>>>(Rb, eln2g + bS, eln2b + bS, X);
    }
    // X now holds encoder memory.

    // ==================== decoder ====================
    dec_embed_k<<<elt(MROWS * DEMB_W), 256, 0, stream>>>(
        de_x, emb_type, emb_barbeat, emb_beat_density, emb_pitch,
        emb_duration, emb_velocity, emb_onset_density, DEMBp);
    gemm(DEMBp, dec_in_W, dec_in_b, nullptr, Y, MROWS, DM, DEMB_W);
    beat_pe_k<<<elt(MROWS * DM), 256, 0, stream>>>(de_x, Y);

    for (int i = 0; i < 8; ++i) {
        const size_t wS = (size_t)i * DM * DM, bS = (size_t)i * DM;
        const size_t w1S = (size_t)i * DM * DFF, b1S = (size_t)i * DFF;
        // causal self attention
        gemm(Y, dWq + wS, dbq + bS, nullptr, Qb, MROWS, DM, DM);
        gemm(Y, dWk + wS, dbk + bS, nullptr, Kb, MROWS, DM, DM);
        gemm(Y, dWv + wS, dbv + bS, nullptr, Vb, MROWS, DM, DM);
        fm_k<<<elt(MROWS * DM), 256, 0, stream>>>(Qb, MROWS * DM);
        fm_k<<<elt(MROWS * DM), 256, 0, stream>>>(Kb, MROWS * DM);
        causal_linattn_k<<<BB * NH, 64, 0, stream>>>(Qb, Kb, Vb, Ab);
        gemm(Ab, dWo + wS, dbo + bS, Y, Rb, MROWS, DM, DM);
        layernorm_k<<<MROWS, 256, 0, stream>>>(Rb, dln1g + bS, dln1b + bS, Y);
        // cross attention (K,V from encoder memory X)
        gemm(Y, dWq2 + wS, dbq2 + bS, nullptr, Qb, MROWS, DM, DM);
        gemm(X, dWk2 + wS, dbk2 + bS, nullptr, Kb, MROWS, DM, DM);
        gemm(X, dWv2 + wS, dbv2 + bS, nullptr, Vb, MROWS, DM, DM);
        fm_k<<<elt(MROWS * DM), 256, 0, stream>>>(Qb, MROWS * DM);
        fm_k<<<elt(MROWS * DM), 256, 0, stream>>>(Kb, MROWS * DM);
        linattn_k<<<BB * NH, 256, 0, stream>>>(Qb, Kb, Vb, Ab, LL);
        gemm(Ab, dWo2 + wS, dbo2 + bS, Y, Rb, MROWS, DM, DM);
        layernorm_k<<<MROWS, 256, 0, stream>>>(Rb, dln2g + bS, dln2b + bS, Y);
        // FFN
        gemm(Y, dW1 + w1S, db1 + b1S, nullptr, Fb, MROWS, DFF, DM);
        gelu_k<<<elt(MROWS * DFF), 256, 0, stream>>>(Fb, MROWS * DFF);
        gemm(Fb, dW2 + w1S, db2 + bS, Y, Rb, MROWS, DM, DFF);
        layernorm_k<<<MROWS, 256, 0, stream>>>(Rb, dln3g + bS, dln3b + bS, Y);
    }

    // ==================== outputs: (y, y_type) concatenated ====================
    hipMemcpyAsync(d_out, Y, (size_t)MROWS * DM * sizeof(float),
                   hipMemcpyDeviceToDevice, stream);
    proj_type_k<<<elt(MROWS * 4), 256, 0, stream>>>(
        Y, proj_W, proj_b, (float*)d_out + (size_t)MROWS * DM);
}
